// SequenceTagger_40973988004504
// MI455X (gfx1250) — compile-verified, matrix operand order checked by reading, not run on hardware
//
#include <hip/hip_runtime.h>

// ---------------------------------------------------------------------------
// Types for CDNA5 WMMA (wave32): bf16 16x16x32, f32 accumulate
// ---------------------------------------------------------------------------
typedef __attribute__((ext_vector_type(16))) __bf16 v16bf;
typedef __attribute__((ext_vector_type(8)))  float  v8f;
typedef __attribute__((ext_vector_type(4)))  unsigned int u32x4;

union FragU {
    u32x4 u4[2];   // two 16-byte chunks per lane (A/B fragment layout)
    v16bf v;
};

__device__ __forceinline__ v8f wmma_bf16(const FragU& a, const FragU& b, v8f c) {
    return __builtin_amdgcn_wmma_f32_16x16x32_bf16(
        /*neg_a=*/false, a.v, /*neg_b=*/false, b.v,
        /*c_mod=*/(short)0, c, /*reuse_a=*/false, /*reuse_b=*/false);
}

__device__ __forceinline__ float fast_sigmoid(float x) {
    return 1.f / (1.f + __expf(-x));
}
__device__ __forceinline__ float fast_tanh(float x) {
    return 1.f - 2.f / (__expf(2.f * x) + 1.f);
}

// ---------------------------------------------------------------------------
// fp32 -> bf16 conversion (grid-stride)
// ---------------------------------------------------------------------------
__global__ void cvt_f32_bf16(const float* __restrict__ src,
                             __bf16* __restrict__ dst, int n) {
    int i = blockIdx.x * blockDim.x + threadIdx.x;
    int stride = gridDim.x * blockDim.x;
    for (; i < n; i += stride) dst[i] = (__bf16)src[i];
}

// ---------------------------------------------------------------------------
// GEMM: C[M,N] = A[M,K] * B[N,K]^T + bias[N]
// A,B bf16; C fp32 or bf16. Block tile 128x128x32, 8 waves, LDS-staged.
// Each wave: 4 (M) x 2 (N) accumulator tiles of 16x16.
// ---------------------------------------------------------------------------
template <bool STORE_BF16>
__global__ __launch_bounds__(256)
void gemm_bf16_nt(const __bf16* __restrict__ A,
                  const __bf16* __restrict__ Bm,
                  const float*  __restrict__ bias,
                  float* __restrict__ Cf, __bf16* __restrict__ Cb,
                  int M, int N, int K) {
    __shared__ __align__(16) __bf16 shA[128 * 32];
    __shared__ __align__(16) __bf16 shB[128 * 32];

    const int tid  = threadIdx.x;
    const int wid  = tid >> 5;      // 0..7
    const int lane = tid & 31;
    const int wm   = wid >> 2;      // 0..1  (64-row strip)
    const int wn   = wid & 3;       // 0..3  (32-col strip)
    const int lr   = lane & 15;
    const int lh   = lane >> 4;
    const int bm   = blockIdx.x * 128;
    const int bn   = blockIdx.y * 128;

    // cooperative staging: thread -> (row, 16-elem chunk)
    const int srow = tid >> 1;         // 0..127
    const int scol = (tid & 1) * 16;   // 0 or 16 elems

    v8f acc[4][2] = {};

#pragma unroll 1
    for (int k0 = 0; k0 < K; k0 += 32) {
        const u32x4* ga = (const u32x4*)(A  + (size_t)(bm + srow) * K + k0 + scol);
        const u32x4* gb = (const u32x4*)(Bm + (size_t)(bn + srow) * K + k0 + scol);
        u32x4 a0 = ga[0], a1 = ga[1];
        u32x4 b0 = gb[0], b1 = gb[1];
        if (k0 + 32 < K) {  // hint next K-slice into cache (global_prefetch)
            __builtin_prefetch(A  + (size_t)(bm + srow) * K + k0 + 32 + scol, 0, 0);
            __builtin_prefetch(Bm + (size_t)(bn + srow) * K + k0 + 32 + scol, 0, 0);
        }
        *(u32x4*)&shA[srow * 32 + scol]     = a0;
        *(u32x4*)&shA[srow * 32 + scol + 8] = a1;
        *(u32x4*)&shB[srow * 32 + scol]     = b0;
        *(u32x4*)&shB[srow * 32 + scol + 8] = b1;
        __syncthreads();

        FragU af[4], bf[2];
#pragma unroll
        for (int mt = 0; mt < 4; ++mt) {
            int row = wm * 64 + mt * 16 + lr;
            af[mt].u4[0] = *(const u32x4*)&shA[row * 32 + 8 * lh];
            af[mt].u4[1] = *(const u32x4*)&shA[row * 32 + 16 + 8 * lh];
        }
#pragma unroll
        for (int nt = 0; nt < 2; ++nt) {
            int row = wn * 32 + nt * 16 + lr;
            bf[nt].u4[0] = *(const u32x4*)&shB[row * 32 + 8 * lh];
            bf[nt].u4[1] = *(const u32x4*)&shB[row * 32 + 16 + 8 * lh];
        }
#pragma unroll
        for (int mt = 0; mt < 4; ++mt)
#pragma unroll
            for (int nt = 0; nt < 2; ++nt)
                acc[mt][nt] = wmma_bf16(af[mt], bf[nt], acc[mt][nt]);
        __syncthreads();
    }

#pragma unroll
    for (int mt = 0; mt < 4; ++mt)
#pragma unroll
        for (int nt = 0; nt < 2; ++nt) {
            int col = bn + wn * 32 + nt * 16 + lr;
            float bv = bias[col];
#pragma unroll
            for (int i = 0; i < 8; ++i) {
                int row = bm + wm * 64 + mt * 16 + i + 8 * lh;
                float v = acc[mt][nt][i] + bv;
                if constexpr (STORE_BF16) Cb[(size_t)row * N + col] = (__bf16)v;
                else                      Cf[(size_t)row * N + col] = v;
            }
        }
}

// ---------------------------------------------------------------------------
// LSTM scan: blockIdx.x = direction. 512 threads = 16 waves; wave w owns the
// two h-column tiles {w, w+16} (columns [16w,16w+16) and [16(w+16), ...)),
// sharing one set of A (h) fragments. h (32x512 bf16) lives in LDS, cell
// state c stays in VGPRs for all 128 steps. 16 waves -> ~256 VGPR budget per
// wave, so the ~210-reg live set (gacc 128 + cst 32 + frags) fits w/o spills.
// ---------------------------------------------------------------------------
__global__ __launch_bounds__(512)
void lstm_scan(const float* __restrict__ xg_f, const float* __restrict__ xg_b,
               const __bf16* __restrict__ whh_f, const __bf16* __restrict__ whh_b,
               __bf16* __restrict__ hcat) {
    __shared__ __align__(16) __bf16 hsh[32 * 512];

    const int dir  = blockIdx.x;
    const float*  xg  = dir ? xg_b  : xg_f;
    const __bf16* whh = dir ? whh_b : whh_f;

    const int tid  = threadIdx.x;
    const int wid  = tid >> 5;   // 0..15
    const int lane = tid & 31;
    const int lr   = lane & 15;
    const int lh   = lane >> 4;

    for (int idx = tid; idx < 32 * 512; idx += 512) hsh[idx] = (__bf16)0.f;
    v8f cst[2][2] = {};   // [jj][mt]
    __syncthreads();

    for (int s = 0; s < 128; ++s) {
        const int t = dir ? (127 - s) : s;
        v8f gacc[2][4][2] = {};   // [jj][gate][mt]

#pragma unroll 1
        for (int k0 = 0; k0 < 512; k0 += 32) {
            FragU af[2];
#pragma unroll
            for (int mt = 0; mt < 2; ++mt) {
                int row = mt * 16 + lr;   // batch row
                af[mt].u4[0] = *(const u32x4*)&hsh[row * 512 + k0 + 8 * lh];
                af[mt].u4[1] = *(const u32x4*)&hsh[row * 512 + k0 + 16 + 8 * lh];
            }
#pragma unroll
            for (int g = 0; g < 4; ++g) {
#pragma unroll
                for (int jj = 0; jj < 2; ++jj) {
                    FragU bf;
                    int nrow = g * 512 + (wid + 16 * jj) * 16 + lr;  // W_hh row
                    bf.u4[0] = *(const u32x4*)&whh[(size_t)nrow * 512 + k0 + 8 * lh];
                    bf.u4[1] = *(const u32x4*)&whh[(size_t)nrow * 512 + k0 + 16 + 8 * lh];
#pragma unroll
                    for (int mt = 0; mt < 2; ++mt)
                        gacc[jj][g][mt] = wmma_bf16(af[mt], bf, gacc[jj][g][mt]);
                }
            }
        }
        __syncthreads();   // all reads of h done before overwrite

#pragma unroll
        for (int jj = 0; jj < 2; ++jj) {
            const int col = (wid + 16 * jj) * 16 + lr;
#pragma unroll
            for (int mt = 0; mt < 2; ++mt) {
#pragma unroll
                for (int i = 0; i < 8; ++i) {
                    int batch = mt * 16 + i + 8 * lh;
                    size_t xb = ((size_t)t * 32 + batch) * 2048 + col;
                    float gi = gacc[jj][0][mt][i] + xg[xb + 0 * 512];
                    float gf = gacc[jj][1][mt][i] + xg[xb + 1 * 512];
                    float gg = gacc[jj][2][mt][i] + xg[xb + 2 * 512];
                    float go = gacc[jj][3][mt][i] + xg[xb + 3 * 512];
                    float iv = fast_sigmoid(gi);
                    float fv = fast_sigmoid(gf);
                    float gv = fast_tanh(gg);
                    float ov = fast_sigmoid(go);
                    float c  = fv * cst[jj][mt][i] + iv * gv;
                    cst[jj][mt][i] = c;
                    float h  = ov * fast_tanh(c);
                    hsh[batch * 512 + col] = (__bf16)h;
                    hcat[((size_t)t * 32 + batch) * 1024 + dir * 512 + col] = (__bf16)h;
                }
            }
        }
        __syncthreads();   // h fully written for next step
    }
}

// ---------------------------------------------------------------------------
// feats[b,t,k] = hcat[t*32+b,:] . W_lin[k,:] + b_lin[k]   (K=20, tiny GEMM)
// ---------------------------------------------------------------------------
__global__ void feats_kernel(const __bf16* __restrict__ hcat,
                             const __bf16* __restrict__ wlin,
                             const float*  __restrict__ blin,
                             float* __restrict__ feats) {
    int idx = blockIdx.x * blockDim.x + threadIdx.x;
    if (idx >= 128 * 32 * 20) return;
    int k  = idx % 20;
    int tb = idx / 20;
    int t  = tb >> 5;
    int b  = tb & 31;
    const __bf16* hr = hcat + (size_t)tb * 1024;
    const __bf16* wr = wlin + (size_t)k * 1024;
    float s = blin[k];
    for (int e = 0; e < 1024; e += 8) {
        u32x4 hu = *(const u32x4*)&hr[e];
        u32x4 wu = *(const u32x4*)&wr[e];
        const __bf16* hp = (const __bf16*)&hu;
        const __bf16* wp = (const __bf16*)&wu;
#pragma unroll
        for (int q = 0; q < 8; ++q) s += (float)hp[q] * (float)wp[q];
    }
    feats[((size_t)b * 128 + t) * 20 + k] = s;
}

// ---------------------------------------------------------------------------
// CRF NLL: one wave per batch element (wave32), lane = tag state.
// ---------------------------------------------------------------------------
#define KTAGS 20
#define START_TAG 18
#define STOP_TAG 19
#define NEGV (-10000.0f)

__global__ __launch_bounds__(1024)
void crf_kernel(const float* __restrict__ feats, const int* __restrict__ tags,
                const float* __restrict__ trans, float* __restrict__ out) {
    __shared__ float str[KTAGS * KTAGS];
    int tid = threadIdx.x;
    if (tid < KTAGS * KTAGS) str[tid] = trans[tid];
    __syncthreads();

    int b    = tid >> 5;
    int lane = tid & 31;
    bool act = lane < KTAGS;
    int li   = act ? lane : 0;
    const float* fb = feats + (size_t)b * 128 * KTAGS;

    float alpha = (lane == START_TAG) ? 0.f : NEGV;
    for (int t = 0; t < 128; ++t) {
        float sc[KTAGS];
        float mx = -3.4e38f;
#pragma unroll
        for (int i = 0; i < KTAGS; ++i) {
            float ai = __shfl(alpha, i, 32);
            float v  = ai + str[li * KTAGS + i];
            sc[i] = v;
            mx = fmaxf(mx, v);
        }
        float s = 0.f;
#pragma unroll
        for (int i = 0; i < KTAGS; ++i) s += __expf(sc[i] - mx);
        float na = mx + __logf(s) + fb[t * KTAGS + li];
        alpha = act ? na : NEGV;
    }
    float fin = act ? (alpha + str[STOP_TAG * KTAGS + li]) : -3.4e38f;
    float mx = -3.4e38f;
#pragma unroll
    for (int i = 0; i < KTAGS; ++i) mx = fmaxf(mx, __shfl(fin, i, 32));
    float s = 0.f;
#pragma unroll
    for (int i = 0; i < KTAGS; ++i) s += __expf(__shfl(fin, i, 32) - mx);
    float logZ = mx + __logf(s);

    if (lane == 0) {
        const int* tg = tags + b * 128;
        float tscore = 0.f, escore = 0.f;
        int prev = START_TAG;
        for (int t = 0; t < 128; ++t) {
            int cur = tg[t];
            tscore += str[cur * KTAGS + prev];
            escore += fb[t * KTAGS + cur];
            prev = cur;
        }
        tscore += str[STOP_TAG * KTAGS + prev];
        out[b] = logZ - (tscore + escore);
    }
}

// ---------------------------------------------------------------------------
// Host orchestration
// ---------------------------------------------------------------------------
extern "C" void kernel_launch(void* const* d_in, const int* in_sizes, int n_in,
                              void* d_out, int out_size, void* d_ws, size_t ws_size,
                              hipStream_t stream) {
    const float* x      = (const float*)d_in[0];
    const int*   tags   = (const int*)  d_in[1];
    const float* W_e2n  = (const float*)d_in[2];
    const float* b_e2n  = (const float*)d_in[3];
    const float* W_ih_f = (const float*)d_in[4];
    const float* W_hh_f = (const float*)d_in[5];
    const float* b_f    = (const float*)d_in[6];
    const float* W_ih_b = (const float*)d_in[7];
    const float* W_hh_b = (const float*)d_in[8];
    const float* b_b    = (const float*)d_in[9];
    const float* W_lin  = (const float*)d_in[10];
    const float* b_lin  = (const float*)d_in[11];
    const float* trans  = (const float*)d_in[12];
    float* out = (float*)d_out;

    char* ws = (char*)d_ws;
    const size_t MB = 1024ull * 1024ull;
    // Region A (32MB): x_bf, then reused for xg_f (fp32) after x_bf is dead.
    __bf16* x_bf    = (__bf16*)(ws + 0);
    float*  xg_f    = (float*) (ws + 0);
    // Region B (32MB): W_e2n_bf, then reused for xg_b.
    __bf16* We2n_bf = (__bf16*)(ws + 32 * MB);
    float*  xg_b    = (float*) (ws + 32 * MB);
    __bf16* Xn_bf   = (__bf16*)(ws + 64 * MB);
    __bf16* Wihf_bf = (__bf16*)(ws + 96 * MB);
    __bf16* Wihb_bf = (__bf16*)(ws + 112 * MB);
    __bf16* Whhf_bf = (__bf16*)(ws + 128 * MB);
    __bf16* Whhb_bf = (__bf16*)(ws + 130 * MB);
    __bf16* hcat_bf = (__bf16*)(ws + 132 * MB);
    __bf16* Wlin_bf = (__bf16*)(ws + 140 * MB);
    float*  featsb  = (float*) (ws + 141 * MB);

    auto cvt = [&](const float* s, __bf16* d, int n) {
        int blocks = (n + 255) / 256;
        if (blocks > 8192) blocks = 8192;
        cvt_f32_bf16<<<blocks, 256, 0, stream>>>(s, d, n);
    };
    cvt(x,      x_bf,    4096 * 4096);
    cvt(W_e2n,  We2n_bf, 4096 * 4096);
    cvt(W_ih_f, Wihf_bf, 2048 * 4096);
    cvt(W_ih_b, Wihb_bf, 2048 * 4096);
    cvt(W_hh_f, Whhf_bf, 2048 * 512);
    cvt(W_hh_b, Whhb_bf, 2048 * 512);
    cvt(W_lin,  Wlin_bf, 20 * 1024);

    // Xn = x @ W_e2n^T + b_e2n  (store bf16; only consumed by next GEMMs)
    dim3 g1(32, 32);
    gemm_bf16_nt<true><<<g1, 256, 0, stream>>>(x_bf, We2n_bf, b_e2n,
                                               nullptr, Xn_bf, 4096, 4096, 4096);
    // xg_{f,b} = Xn @ W_ih^T + b  (fp32 for the recurrent accumulation)
    dim3 g2(32, 16);
    gemm_bf16_nt<false><<<g2, 256, 0, stream>>>(Xn_bf, Wihf_bf, b_f,
                                                xg_f, nullptr, 4096, 2048, 4096);
    gemm_bf16_nt<false><<<g2, 256, 0, stream>>>(Xn_bf, Wihb_bf, b_b,
                                                xg_b, nullptr, 4096, 2048, 4096);

    // Sequential bidirectional LSTM scan (2 blocks = 2 directions in parallel)
    lstm_scan<<<2, 512, 0, stream>>>(xg_f, xg_b, Whhf_bf, Whhb_bf, hcat_bf);

    // Output projection K=20 and CRF negative log-likelihood
    feats_kernel<<<(128 * 32 * 20 + 255) / 256, 256, 0, stream>>>(
        hcat_bf, Wlin_bf, b_lin, featsb);
    crf_kernel<<<1, 1024, 0, stream>>>(featsb, tags, trans, out);
}
